// AttentionBlock_11519102288060
// MI455X (gfx1250) — matrix operand order repaired and checked by reading, not compiled, not run on hardware
//
#include <hip/hip_runtime.h>
#include <hip/hip_bf16.h>

typedef __attribute__((ext_vector_type(16))) _Float16 v16h;
typedef __attribute__((ext_vector_type(8)))  _Float16 v8h;
typedef __attribute__((ext_vector_type(8)))  float    v8f;
typedef __attribute__((ext_vector_type(4)))  int      v4i;

// ---------------------------------------------------------------------------
// Optional CDNA5 async global->LDS path (ASYNCcnt-tracked DMA-style copies).
// Guarded: falls back to global->VGPR->ds_store staging if builtins absent.
// Builtin signature (from hipcc diagnostic): (global int4*, lds int4*, Ii, Ii).
// ---------------------------------------------------------------------------
#if defined(__has_builtin)
#  if __has_builtin(__builtin_amdgcn_global_load_async_to_lds_b128) && \
      __has_builtin(__builtin_amdgcn_s_wait_asynccnt)
#    define HAVE_ASYNC_LDS 1
#  endif
#endif
#ifndef HAVE_ASYNC_LDS
#  define HAVE_ASYNC_LDS 0
#endif

#if HAVE_ASYNC_LDS
typedef __attribute__((address_space(1))) v4i* gptr4;
typedef __attribute__((address_space(3))) v4i* lptr4;
__device__ __forceinline__ void async_cp16(const _Float16* g, _Float16* l) {
  __builtin_amdgcn_global_load_async_to_lds_b128((gptr4)(const void*)g,
                                                 (lptr4)(void*)l, 0, 0);
}
__device__ __forceinline__ void async_wait0() {
  __builtin_amdgcn_s_wait_asynccnt(0);
}
#define PANBUF 2
#else
#define PANBUF 1
#endif

// ---------------------------------------------------------------------------
// WMMA helper: D = A(16x32 f16) * B(32x16 f16) + C(16x16 f32)
// ---------------------------------------------------------------------------
__device__ __forceinline__ v8f wmma_f16(v16h a, v16h b, v8f c) {
  return __builtin_amdgcn_wmma_f32_16x16x32_f16(false, a, false, b,
                                                (short)0, c, false, false);
}

// A-fragment from row-major [M][K] memory (ld = K stride, elements).
// Lane L<16: row M=L, K 0..15; lane>=16: row M=L-16, K 16..31.
__device__ __forceinline__ v16h load_a_frag(const _Float16* base, int ld) {
  const int lane = threadIdx.x & 31;
  const int row  = lane & 15;
  const int koff = (lane >> 4) << 4;           // 0 or 16
  const _Float16* p = base + (size_t)row * ld + koff;
  v8h lo = *(const v8h*)(p);
  v8h hi = *(const v8h*)(p + 8);
  v16h r;
#pragma unroll
  for (int i = 0; i < 8; ++i) { r[i] = lo[i]; r[i + 8] = hi[i]; }
  return r;
}

// B-fragment (K x 16) from memory laid out [N][K] with K contiguous (ld = stride
// between N rows). Lane L<16: col N=L, K 0..15; lane>=16: col N=L-16, K 16..31.
__device__ __forceinline__ v16h load_b_frag(const _Float16* base, int ld) {
  const int lane = threadIdx.x & 31;
  const _Float16* p = base + (size_t)(lane & 15) * ld + ((lane >> 4) << 4);
  return *(const v16h*)p;                       // 32B contiguous
}

// ---------------------------------------------------------------------------
// Kernel 1: GroupNorm(32 groups) over [b, 512, 1024]; write f16 transposed
// to xn[b][t][c] so QKV B-fragments are c-contiguous.
// ---------------------------------------------------------------------------
__global__ __launch_bounds__(256) void gn_kernel(const float* __restrict__ x,
                                                 const float* __restrict__ gs,
                                                 const float* __restrict__ gb,
                                                 _Float16* __restrict__ xn) {
  __shared__ float s1[256], s2[256];
  const int tid = threadIdx.x;
  const int b = blockIdx.x >> 5;
  const int g = blockIdx.x & 31;
  const float* xb = x + ((size_t)b * 512 + g * 16) * 1024;

  float sum = 0.f, sum2 = 0.f;
  for (int i = tid; i < 16 * 1024; i += 256) {
    float v = xb[i];
    sum += v; sum2 += v * v;
  }
  s1[tid] = sum; s2[tid] = sum2;
  __syncthreads();
#pragma unroll
  for (int off = 128; off > 0; off >>= 1) {
    if (tid < off) { s1[tid] += s1[tid + off]; s2[tid] += s2[tid + off]; }
    __syncthreads();
  }
  const float mu   = s1[0] * (1.f / 16384.f);
  const float var  = s2[0] * (1.f / 16384.f) - mu * mu;
  const float rstd = rsqrtf(var + 1e-5f);

  for (int i = tid; i < 16 * 1024; i += 256) {
    const int ci = i >> 10;            // 0..15
    const int t  = i & 1023;
    const int c  = g * 16 + ci;
    float v = (xb[i] - mu) * rstd * gs[c] + gb[c];
    xn[((size_t)b * 1024 + t) * 512 + c] = (_Float16)v;
  }
}

// ---------------------------------------------------------------------------
// Kernel 2: weight conversion f32 -> f16
// ---------------------------------------------------------------------------
__global__ __launch_bounds__(256) void cvt_kernel(const float* __restrict__ wq,
                                                  const float* __restrict__ wp,
                                                  _Float16* __restrict__ wq16,
                                                  _Float16* __restrict__ wp16) {
  const int i = blockIdx.x * 256 + threadIdx.x;
  if (i < 1536 * 512) wq16[i] = (_Float16)wq[i];
  if (i < 512 * 512)  wp16[i] = (_Float16)wp[i];
}

// ---------------------------------------------------------------------------
// Kernel 3: QKV GEMM. Out[o,t] = sum_c Wq[o,c] * xn[b][t][c] + bias[o]
// Block: 64-row weight panel staged in LDS (shared by 8 waves, async double-
// buffered when available), each wave computes a 64M x 32N tile.
//   q,k : [bh][t][64] f16 (scaled by 64^-0.25),  v : [bh][64][t] f16
// ---------------------------------------------------------------------------
__device__ __forceinline__ void qkv_store_tile(const v8f d, int obase, int tbase,
                                               int b, const float* __restrict__ bq,
                                               _Float16* __restrict__ qb,
                                               _Float16* __restrict__ kbuf,
                                               _Float16* __restrict__ vbuf) {
  const int lane = threadIdx.x & 31;
  const int mh   = (lane >> 4) << 3;       // 0 or 8
  const int t    = tbase + (lane & 15);
  const int head = obase / 192;
  const int rr   = obase % 192;
  const int kind = rr >> 6;                // 0=q 1=k 2=v
  const int c0   = (rr & 63) + mh;
  const int bh   = b * 8 + head;
  if (kind == 2) {
#pragma unroll
    for (int r = 0; r < 8; ++r) {
      float val = d[r] + bq[obase + mh + r];
      vbuf[((size_t)bh * 64 + c0 + r) * 1024 + t] = (_Float16)val;
    }
  } else {
    const float sc = 0.35355339059327373f;  // 64^-0.25
    _Float16* dst = (kind == 0 ? qb : kbuf) + ((size_t)(bh * 1024 + t) * 64 + c0);
    v8h out;
#pragma unroll
    for (int r = 0; r < 8; ++r) out[r] = (_Float16)((d[r] + bq[obase + mh + r]) * sc);
    *(v8h*)dst = out;
  }
}

__device__ __forceinline__ void gemm_inner128(v8f d[8], const _Float16* pan,
                                              const _Float16* xrow0, int kofs) {
#pragma unroll
  for (int kk = 0; kk < 128; kk += 32) {
    v16h a[4];
#pragma unroll
    for (int mi = 0; mi < 4; ++mi)
      a[mi] = load_a_frag(pan + mi * 16 * 128 + kk, 128);
    v16h b0 = load_b_frag(xrow0 + kofs + kk, 512);
    v16h b1 = load_b_frag(xrow0 + 16 * 512 + kofs + kk, 512);
#pragma unroll
    for (int mi = 0; mi < 4; ++mi) {
      d[mi * 2 + 0] = wmma_f16(a[mi], b0, d[mi * 2 + 0]);
      d[mi * 2 + 1] = wmma_f16(a[mi], b1, d[mi * 2 + 1]);
    }
  }
}

__global__ __launch_bounds__(256) void qkv_gemm_kernel(const _Float16* __restrict__ wq,
                                                       const _Float16* __restrict__ xn,
                                                       const float* __restrict__ bq,
                                                       _Float16* __restrict__ qb,
                                                       _Float16* __restrict__ kbuf,
                                                       _Float16* __restrict__ vbuf) {
  __shared__ __align__(32) _Float16 apan[PANBUF][64 * 128];   // 16/32 KB
  const int tid  = threadIdx.x;
  const int wave = tid >> 5;
  const int b  = blockIdx.y;
  const int m0 = (blockIdx.x >> 2) * 64;              // 24 m-blocks
  const int n0 = (blockIdx.x & 3) * 256 + wave * 32;  // 4 n-blocks x 8 waves
  const _Float16* xrow0 = xn + (size_t)b * 1024 * 512 + (size_t)n0 * 512;
  const int prow = tid >> 2, pc8 = (tid & 3) * 32;    // 64 rows x 4 chunks(8h) x4 its

  v8f d[8] = {};
#if HAVE_ASYNC_LDS
#pragma unroll
  for (int c = 0; c < 4; ++c)
    async_cp16(wq + (size_t)(m0 + prow) * 512 + pc8 + c * 8,
               &apan[0][prow * 128 + pc8 + c * 8]);
  async_wait0();
  __syncthreads();
  for (int ki = 0; ki < 4; ++ki) {
    const int cur = ki & 1, nxt = cur ^ 1;
    const int kpre = ((ki + 1) & 3) * 128;            // wraps; last one harmless
#pragma unroll
    for (int c = 0; c < 4; ++c)
      async_cp16(wq + (size_t)(m0 + prow) * 512 + kpre + pc8 + c * 8,
                 &apan[nxt][prow * 128 + pc8 + c * 8]);
    gemm_inner128(d, apan[cur], xrow0, ki * 128);
    async_wait0();
    __syncthreads();
  }
#else
  for (int kblk = 0; kblk < 512; kblk += 128) {
    __syncthreads();
#pragma unroll
    for (int c = 0; c < 4; ++c)
      *(v8h*)&apan[0][prow * 128 + pc8 + c * 8] =
          *(const v8h*)(wq + (size_t)(m0 + prow) * 512 + kblk + pc8 + c * 8);
    __syncthreads();
    gemm_inner128(d, apan[0], xrow0, kblk);
  }
#endif
#pragma unroll
  for (int mi = 0; mi < 4; ++mi) {
    qkv_store_tile(d[mi * 2 + 0], m0 + mi * 16, n0,      b, bq, qb, kbuf, vbuf);
    qkv_store_tile(d[mi * 2 + 1], m0 + mi * 16, n0 + 16, b, bq, qb, kbuf, vbuf);
  }
}

// ---------------------------------------------------------------------------
// Kernel 4: flash attention. Block = one (b,h), 8 waves x 16 query rows.
// K/V tiles (32x64 / 64x32 f16) double-buffered in LDS (async staged when
// available); S = Qt x K, online softmax (f32, shfl reductions matched to the
// C/D VGPR layout), P transposed D->A layout through LDS, O += P x V.
// ---------------------------------------------------------------------------
__global__ __launch_bounds__(256) void attn_kernel(const _Float16* __restrict__ qg,
                                                   const _Float16* __restrict__ kg,
                                                   const _Float16* __restrict__ vg,
                                                   _Float16* __restrict__ ag) {
  __shared__ __align__(32) _Float16 kt[2][32 * 64];   // 4 KB x2
  __shared__ __align__(32) _Float16 vt[2][64 * 32];   // 4 KB x2
  __shared__ __align__(32) _Float16 plds[8][16 * 32]; // 8 KB
  const int tid  = threadIdx.x;
  const int wave = tid >> 5;
  const int lane = tid & 31;
  const int bh = blockIdx.x >> 3;
  const int t0 = (blockIdx.x & 7) * 128 + wave * 16;
  const int b = bh >> 3, h = bh & 7;

  const _Float16* kbh = kg + (size_t)bh * 1024 * 64;
  const _Float16* vbh = vg + (size_t)bh * 64 * 1024;
  const int krow = tid >> 3, kc8 = (tid & 7) * 8;     // 32 rows x 8 chunks
  const int vrow = tid >> 2, vc8 = (tid & 3) * 8;     // 64 rows x 4 chunks

  const _Float16* qbase = qg + (size_t)(bh * 1024 + t0) * 64;
  const v16h qa0 = load_a_frag(qbase, 64);
  const v16h qa1 = load_a_frag(qbase + 32, 64);

  float mrun[8], lrun[8];
  v8f acc[4] = {};
#pragma unroll
  for (int r = 0; r < 8; ++r) { mrun[r] = -1e30f; lrun[r] = 0.f; }

  // pre-stage s0 = 0 into buffer 0
#if HAVE_ASYNC_LDS
  async_cp16(kbh + (size_t)krow * 64 + kc8, &kt[0][krow * 64 + kc8]);
  async_cp16(vbh + (size_t)vrow * 1024 + vc8, &vt[0][vrow * 32 + vc8]);
  async_wait0();
#else
  *(v8h*)&kt[0][krow * 64 + kc8] = *(const v8h*)(kbh + (size_t)krow * 64 + kc8);
  *(v8h*)&vt[0][vrow * 32 + vc8] = *(const v8h*)(vbh + (size_t)vrow * 1024 + vc8);
#endif
  __syncthreads();

  for (int it = 0; it < 32; ++it) {
    const int cur = it & 1, nxt = cur ^ 1;
    const int sn = ((it + 1) & 31) * 32;   // wraps; last prefetch is harmless
#if HAVE_ASYNC_LDS
    async_cp16(kbh + (size_t)(sn + krow) * 64 + kc8, &kt[nxt][krow * 64 + kc8]);
    async_cp16(vbh + (size_t)vrow * 1024 + sn + vc8, &vt[nxt][vrow * 32 + vc8]);
#else
    v8h kn = *(const v8h*)(kbh + (size_t)(sn + krow) * 64 + kc8);
    v8h vn = *(const v8h*)(vbh + (size_t)vrow * 1024 + sn + vc8);
#endif

    v8f st0 = {}, st1 = {};
    st0 = wmma_f16(qa0, load_b_frag(&kt[cur][0], 64), st0);
    st0 = wmma_f16(qa1, load_b_frag(&kt[cur][32], 64), st0);
    st1 = wmma_f16(qa0, load_b_frag(&kt[cur][16 * 64], 64), st1);
    st1 = wmma_f16(qa1, load_b_frag(&kt[cur][16 * 64 + 32], 64), st1);

    float corr[8];
#pragma unroll
    for (int r = 0; r < 8; ++r) {
      float mx = fmaxf(st0[r], st1[r]);
#pragma unroll
      for (int msk = 8; msk >= 1; msk >>= 1) mx = fmaxf(mx, __shfl_xor(mx, msk, 32));
      const float mn = fmaxf(mrun[r], mx);
      corr[r] = __expf(mrun[r] - mn);
      const float p0 = __expf(st0[r] - mn);
      const float p1 = __expf(st1[r] - mn);
      float ps = p0 + p1;
#pragma unroll
      for (int msk = 8; msk >= 1; msk >>= 1) ps += __shfl_xor(ps, msk, 32);
      lrun[r] = lrun[r] * corr[r] + ps;
      mrun[r] = mn;
      const int mrow = r + ((lane >> 4) << 3);
      plds[wave][mrow * 32 + (lane & 15)]      = (_Float16)p0;
      plds[wave][mrow * 32 + 16 + (lane & 15)] = (_Float16)p1;
    }
#pragma unroll
    for (int j = 0; j < 4; ++j)
#pragma unroll
      for (int r = 0; r < 8; ++r) acc[j][r] *= corr[r];

    // reload P as an A-fragment (16 rows x 32 s) from per-wave LDS
    v16h pa;
    {
      const int row  = lane & 15;
      const int koff = (lane >> 4) << 4;
      const _Float16* p = &plds[wave][row * 32 + koff];
      v8h lo = *(const v8h*)p;
      v8h hi = *(const v8h*)(p + 8);
#pragma unroll
      for (int i = 0; i < 8; ++i) { pa[i] = lo[i]; pa[i + 8] = hi[i]; }
    }
#pragma unroll
    for (int j = 0; j < 4; ++j)
      acc[j] = wmma_f16(pa, load_b_frag(&vt[cur][j * 16 * 32], 32), acc[j]);

#if HAVE_ASYNC_LDS
    async_wait0();
#else
    *(v8h*)&kt[nxt][krow * 64 + kc8] = kn;
    *(v8h*)&vt[nxt][vrow * 32 + vc8] = vn;
#endif
    __syncthreads();
  }

  float inv[8];
#pragma unroll
  for (int r = 0; r < 8; ++r) inv[r] = 1.0f / lrun[r];
  const int mh = (lane >> 4) << 3;
#pragma unroll
  for (int j = 0; j < 4; ++j) {
    const int cg = h * 64 + j * 16 + (lane & 15);
#pragma unroll
    for (int r = 0; r < 8; ++r) {
      const int t = t0 + mh + r;
      ag[((size_t)b * 1024 + t) * 512 + cg] = (_Float16)(acc[j][r] * inv[r]);
    }
  }
}

// ---------------------------------------------------------------------------
// Kernel 5: proj GEMM + bias + residual, same blocking as QKV GEMM.
// out[b][o][t] = x + Wp*a + bp
// ---------------------------------------------------------------------------
__global__ __launch_bounds__(256) void proj_gemm_kernel(const _Float16* __restrict__ wp,
                                                        const _Float16* __restrict__ ab,
                                                        const float* __restrict__ bp,
                                                        const float* __restrict__ x,
                                                        float* __restrict__ out) {
  __shared__ __align__(32) _Float16 apan[PANBUF][64 * 128];
  const int tid  = threadIdx.x;
  const int wave = tid >> 5;
  const int b  = blockIdx.y;
  const int m0 = (blockIdx.x >> 2) * 64;              // 8 m-blocks
  const int n0 = (blockIdx.x & 3) * 256 + wave * 32;  // 4 n-blocks x 8 waves
  const _Float16* arow0 = ab + (size_t)b * 1024 * 512 + (size_t)n0 * 512;
  const int prow = tid >> 2, pc8 = (tid & 3) * 32;

  v8f d[8] = {};
#if HAVE_ASYNC_LDS
#pragma unroll
  for (int c = 0; c < 4; ++c)
    async_cp16(wp + (size_t)(m0 + prow) * 512 + pc8 + c * 8,
               &apan[0][prow * 128 + pc8 + c * 8]);
  async_wait0();
  __syncthreads();
  for (int ki = 0; ki < 4; ++ki) {
    const int cur = ki & 1, nxt = cur ^ 1;
    const int kpre = ((ki + 1) & 3) * 128;
#pragma unroll
    for (int c = 0; c < 4; ++c)
      async_cp16(wp + (size_t)(m0 + prow) * 512 + kpre + pc8 + c * 8,
                 &apan[nxt][prow * 128 + pc8 + c * 8]);
    gemm_inner128(d, apan[cur], arow0, ki * 128);
    async_wait0();
    __syncthreads();
  }
#else
  for (int kblk = 0; kblk < 512; kblk += 128) {
    __syncthreads();
#pragma unroll
    for (int c = 0; c < 4; ++c)
      *(v8h*)&apan[0][prow * 128 + pc8 + c * 8] =
          *(const v8h*)(wp + (size_t)(m0 + prow) * 512 + kblk + pc8 + c * 8);
    __syncthreads();
    gemm_inner128(d, apan[0], arow0, kblk);
  }
#endif

  const int lane = threadIdx.x & 31;
  const int mh   = (lane >> 4) << 3;
#pragma unroll
  for (int mi = 0; mi < 4; ++mi)
#pragma unroll
    for (int ni = 0; ni < 2; ++ni) {
      const int ob = m0 + mi * 16;
      const int t  = n0 + ni * 16 + (lane & 15);
#pragma unroll
      for (int r = 0; r < 8; ++r) {
        const int o = ob + mh + r;
        const size_t idx = ((size_t)b * 512 + o) * 1024 + t;
        out[idx] = x[idx] + d[mi * 2 + ni][r] + bp[o];
      }
    }
}

// ---------------------------------------------------------------------------
// Launch
// ---------------------------------------------------------------------------
extern "C" void kernel_launch(void* const* d_in, const int* in_sizes, int n_in,
                              void* d_out, int out_size, void* d_ws, size_t ws_size,
                              hipStream_t stream) {
  (void)in_sizes; (void)n_in; (void)out_size; (void)ws_size;
  const float* x   = (const float*)d_in[0];
  const float* gs  = (const float*)d_in[1];
  const float* gb  = (const float*)d_in[2];
  const float* wq  = (const float*)d_in[3];
  const float* bq  = (const float*)d_in[4];
  const float* wp  = (const float*)d_in[5];
  const float* bp  = (const float*)d_in[6];
  float* out = (float*)d_out;

  char* ws = (char*)d_ws;
  _Float16* xn   = (_Float16*)(ws);                        // 8 MB  [b][t][512]
  _Float16* wq16 = (_Float16*)(ws + 8388608);              // 1.5MB [1536][512]
  _Float16* wp16 = (_Float16*)(ws + 9961472);              // 0.5MB [512][512]
  _Float16* qbuf = (_Float16*)(ws + 10485760);             // 8 MB  [bh][t][64]
  _Float16* kbuf = (_Float16*)(ws + 18874368);             // 8 MB  [bh][t][64]
  _Float16* vbuf = (_Float16*)(ws + 27262976);             // 8 MB  [bh][64][t]
  _Float16* abuf = (_Float16*)(ws + 35651584);             // 8 MB  [b][t][512]

  gn_kernel<<<256, 256, 0, stream>>>(x, gs, gb, xn);
  cvt_kernel<<<(1536 * 512 + 255) / 256, 256, 0, stream>>>(wq, wp, wq16, wp16);
  qkv_gemm_kernel<<<dim3(96, 8), 256, 0, stream>>>(wq16, xn, bq, qbuf, kbuf, vbuf);
  attn_kernel<<<512, 256, 0, stream>>>(qbuf, kbuf, vbuf, abuf);
  proj_gemm_kernel<<<dim3(32, 8), 256, 0, stream>>>(wp16, abuf, bp, x, out);
}
